// PMNN_8735963480103
// MI455X (gfx1250) — matrix operand order repaired and checked by reading, not compiled, not run on hardware
//
#include <hip/hip_runtime.h>
#include <hip/hip_bf16.h>

// ---------------- CDNA5 WMMA types ----------------
typedef __attribute__((ext_vector_type(16))) _Float16 v16h;
typedef __attribute__((ext_vector_type(8)))  _Float16 v8h;
typedef __attribute__((ext_vector_type(8)))  float    v8f;

// ---------------- constants ----------------
#define SQC      1.0954451150103321f   // sqrt(1.2)
#define CCUR     1.2f
#define MINN     1e-15f
#define MAXN     0.9090672135742957f   // (1 - 4e-3) / sqrt(1.2)
#define AS       136                   // half stride (16B aligned rows)
#define GS       132                   // float stride
#define ROWS     32                    // rows per block chunk

__device__ __forceinline__ float artanh_c(float x) {
    x = fminf(fmaxf(x, -1.0f + 1e-7f), 1.0f - 1e-7f);
    return 0.5f * logf((1.0f + x) / (1.0f - x));
}
__device__ __forceinline__ float tanh_c(float x) {
    return tanhf(fminf(fmaxf(x, -15.0f), 15.0f));
}

// -------- weight pre-pad/convert: W (Nout x Kin, f32) -> Wp (Npad x Kpad, f16, zero-pad) --------
__global__ void __launch_bounds__(256)
convw_kernel(const float* __restrict__ W, _Float16* __restrict__ Wp,
             int Nout, int Kin, int Npad, int Kpad)
{
    const int i = blockIdx.x * 256 + threadIdx.x;
    if (i < Npad * Kpad) {
        const int n = i / Kpad, k = i % Kpad;
        const float v = (n < Nout && k < Kin) ? W[n * Kin + k] : 0.0f;
        Wp[i] = (_Float16)v;
    }
}

// ---------------- WMMA GEMM: sOut(32 x NOUT) = f16(sIn) @ Wp^T ----------------
// Fully unrolled (compile-time shapes) so the scheduler can clause all fragment
// loads ahead of the WMMA chain and use partial s_wait counts.
// A-frag (16-bit 16x32 per ISA): lane L holds row M=L%16; K window depends on L>=16.
// B-frag: lane L holds row N=L%16+16*nt of padded Wp; K window matches A (8*(L>=16) + {0..7, 16..23}) + 32*ks.
template<int NOUT, int KPAD, int KSTEPS>
__device__ __forceinline__ void gemm_f16(const _Float16* __restrict__ sIn,
                                         const _Float16* __restrict__ Wp,
                                         float* __restrict__ sOut)
{
    constexpr int NTILES = (NOUT + 15) / 16;
    constexpr int NJOBS  = 2 * NTILES;          // 2 row tiles x NTILES
    const int tid  = threadIdx.x;
    const int wave = tid >> 5;
    const int lane = tid & 31;
    const int l15  = lane & 15;
    const int lhi  = lane >> 4;                 // 0 or 1

    #pragma unroll
    for (int jb = 0; jb < 2; ++jb) {            // at most 2 jobs per wave (NJOBS <= 16)
        const int job = wave + jb * 8;
        if (job < NJOBS) {
            const int rt = job / NTILES;        // uniform per wave; const divisor
            const int nt = job - rt * NTILES;
            const int n  = l15 + 16 * nt;
            const int m  = l15 + 16 * rt;
            v8f acc = {};
            #pragma unroll
            for (int ks = 0; ks < KSTEPS; ++ks) {
                // ---- A fragment: two contiguous 16B chunks from LDS ----
                const int kh = ks * 32 + lhi * 8;
                const v8h a0 = *(const v8h*)(sIn + m * AS + kh);
                const v8h a1 = *(const v8h*)(sIn + m * AS + kh + 16);
                v16h a;
                #pragma unroll
                for (int j = 0; j < 8; ++j) { a[j] = a0[j]; a[8 + j] = a1[j]; }
                // ---- B fragment: two contiguous 16B chunks from padded global f16 ----
                const int kb = ks * 32 + lhi * 8;
                const v8h b0 = *(const v8h*)(Wp + n * KPAD + kb);
                const v8h b1 = *(const v8h*)(Wp + n * KPAD + kb + 16);
                v16h b;
                #pragma unroll
                for (int j = 0; j < 8; ++j) { b[j] = b0[j]; b[8 + j] = b1[j]; }
                acc = __builtin_amdgcn_wmma_f32_16x16x32_f16(
                    false, a, false, b, (short)0, acc, false, false);
            }
            if (n < NOUT) {                     // folds away when NOUT % 16 == 0
                #pragma unroll
                for (int r2 = 0; r2 < 8; ++r2) {
                    const int mm = r2 + lhi * 8 + 16 * rt;
                    sOut[mm * GS + n] = acc[r2];
                }
            }
        }
    }
}

// expmap0 of a bias vector into LDS (threads t < Nout; redundant norm is cheap)
__device__ __forceinline__ void ebias_stage(const float* __restrict__ bg, int Nout,
                                            float* __restrict__ sEB)
{
    const int t = threadIdx.x;
    if (t < Nout) {
        float s = 0.0f;
        for (int j = 0; j < Nout; ++j) s += bg[j] * bg[j];
        const float n  = fmaxf(sqrtf(s), MINN);
        float fn = tanh_c(SQC * n) / SQC;
        float sc = fn / n;
        if (fn > MAXN) sc *= MAXN / fn;
        sEB[t] = sc * bg[t];
    }
}

// Finish mobius_matvec + mobius_add(eb) + projx for row r.
// Leaves o_j (pre projx scale) in sG row; returns projx'd norm nn; s2 = projx scale.
__device__ __forceinline__ float plinear_core(float* __restrict__ sG, int r, int Nout,
                                              const float* __restrict__ sEB,
                                              float xn, float& s2_out)
{
    float mxn2 = 0.0f; bool allz = true;
    for (int j = 0; j < Nout; ++j) {
        const float v = sG[r * GS + j];
        mxn2 += v * v;
        allz = allz && (v == 0.0f);
    }
    const float mxn = fmaxf(sqrtf(mxn2), MINN);
    float f = tanh_c((mxn / xn) * artanh_c(SQC * xn)) / (mxn * SQC);
    if (allz) f = 0.0f;
    const float rn = fabsf(f) * mxn;          // ||res||
    float s1 = 1.0f;
    if (rn > MAXN) s1 = MAXN / rn;            // projx of matvec result
    const float fx = f * s1;
    float xy = 0.0f, y2 = 0.0f;
    for (int j = 0; j < Nout; ++j) {
        const float e = sEB[j];
        xy += sG[r * GS + j] * e;
        y2 += e * e;
    }
    xy *= fx;
    const float x2  = fx * fx * mxn2;
    const float Am  = 1.0f + 2.0f * CCUR * xy + CCUR * y2;
    const float Bm  = 1.0f - CCUR * x2;
    const float den = fmaxf(1.0f + 2.0f * CCUR * xy + CCUR * CCUR * x2 * y2, MINN);
    float on2 = 0.0f;
    for (int j = 0; j < Nout; ++j) {
        const float o = (Am * fx * sG[r * GS + j] + Bm * sEB[j]) / den;
        sG[r * GS + j] = o;
        on2 += o * o;
    }
    const float on = fmaxf(sqrtf(on2), MINN);
    s2_out = (on > MAXN) ? (MAXN / on) : 1.0f;
    return fminf(on, MAXN);
}

// ---------------- fused main kernel: one 32-row chunk of one batch ----------------
__global__ void __launch_bounds__(256)
hypnet_chunk_kernel(const float* __restrict__ x,
                    const _Float16* __restrict__ We1p, const float* __restrict__ be1,
                    const _Float16* __restrict__ We2p, const float* __restrict__ be2,
                    const _Float16* __restrict__ We3p, const float* __restrict__ be3,
                    const _Float16* __restrict__ Wh1p, const float* __restrict__ bh1,
                    const _Float16* __restrict__ Wh2p, const float* __restrict__ bh2,
                    const _Float16* __restrict__ Wh3p, const float* __restrict__ bh3,
                    const _Float16* __restrict__ Waep, const float* __restrict__ bae,
                    const _Float16* __restrict__ Wahp, const float* __restrict__ bah,
                    const float* __restrict__ Wte, const float* __restrict__ bte,
                    const float* __restrict__ Wth, const float* __restrict__ bth,
                    float* __restrict__ wsE, float* __restrict__ wsH)
{
    __shared__ __align__(16) _Float16 sA[ROWS * AS];   // GEMM input / tan_h (aliased)
    __shared__ __align__(16) _Float16 sE[ROWS * AS];   // out_e (f16)
    __shared__ __align__(16) float    sG[ROWS * GS];   // GEMM f32 output (in-place stages)
    __shared__ __align__(16) float    sH[ROWS * GS];   // out_h (f32)
    __shared__ __align__(16) float    sXS[ROWS * 32];  // xs chunk
    __shared__ float sEB[128];
    __shared__ float sRN[ROWS], sNH[ROWS], sW0[ROWS], sW1[ROWS], sFH[ROWS];

    const int tid = threadIdx.x;
    const int b   = blockIdx.x >> 2;
    const int rc  = blockIdx.x & 3;
    const int r   = tid;

    __builtin_prefetch(We1p, 0, 1);
    __builtin_prefetch(Wh1p, 0, 1);

    // S0: load xs chunk (x is (B,32,128); xs[s][c] = x[b][c][s])
    for (int idx = tid; idx < 32 * ROWS; idx += 256) {
        const int c  = idx >> 5;
        const int rr = idx & 31;
        const float v = x[b * 4096 + c * 128 + rc * ROWS + rr];
        sXS[rr * 32 + c] = v;
        sA[rr * AS + c]  = (_Float16)v;
    }
    __syncthreads();

    // ---------------- Euclidean branch ----------------
    gemm_f16<80, 32, 1>(sA, We1p, sG);  __syncthreads();
    for (int idx = tid; idx < ROWS * 128; idx += 256) {     // relu + zero-pad (cols 80..127)
        const int rr = idx >> 7, c = idx & 127;
        const int cc = min(c, 79);
        float v = fmaxf(sG[rr * GS + cc] + be1[cc], 0.0f);
        v = (c < 80) ? v : 0.0f;
        sA[rr * AS + c] = (_Float16)v;
    }
    __syncthreads();
    gemm_f16<104, 96, 3>(sA, We2p, sG); __syncthreads();
    for (int idx = tid; idx < ROWS * 128; idx += 256) {     // relu + zero-pad (cols 104..127)
        const int rr = idx >> 7, c = idx & 127;
        const int cc = min(c, 103);
        float v = fmaxf(sG[rr * GS + cc] + be2[cc], 0.0f);
        v = (c < 104) ? v : 0.0f;
        sA[rr * AS + c] = (_Float16)v;
    }
    __syncthreads();
    gemm_f16<128, 128, 4>(sA, We3p, sG); __syncthreads();
    for (int idx = tid; idx < ROWS * 128; idx += 256) {     // out_e = relu(. + be3)
        const int rr = idx >> 7, c = idx & 127;
        sE[rr * AS + c] = (_Float16)fmaxf(sG[rr * GS + c] + be3[c], 0.0f);
    }
    __syncthreads();

    // ---------------- Poincare branch ----------------
    // xh = expmap0(xs)
    if (r < ROWS) {
        float s = 0.0f;
        for (int c = 0; c < 32; ++c) { const float v = sXS[r * 32 + c]; s += v * v; }
        const float n = fmaxf(sqrtf(s), MINN);
        float fn = tanh_c(SQC * n) / SQC;
        float sc = fn / n;
        if (fn > MAXN) { sc *= MAXN / fn; fn = MAXN; }
        for (int c = 0; c < 32; ++c) sA[r * AS + c] = (_Float16)(sc * sXS[r * 32 + c]);
        sRN[r] = fmaxf(fn, MINN);
    }
    __syncthreads();

    // H layer 1: p_linear + mob_relu
    gemm_f16<80, 32, 1>(sA, Wh1p, sG); __syncthreads();
    ebias_stage(bh1, 80, sEB);         __syncthreads();
    if (r < ROWS) {
        float s2; const float nn = plinear_core(sG, r, 80, sEB, sRN[r], s2);
        const float ls = artanh_c(SQC * nn) / (SQC * nn);
        float pr2 = 0.0f;
        for (int j = 0; j < 80; ++j) { const float o = sG[r * GS + j]; if (o > 0.0f) pr2 += o * o; }
        const float un = fmaxf(ls * s2 * sqrtf(pr2), MINN);
        const float es = tanh_c(SQC * un) / (SQC * un);
        const float fn = es * un;                       // tanh(SQC*un)/SQC
        const float s3 = (fn > MAXN) ? (MAXN / fn) : 1.0f;
        const float g  = s3 * es * ls * s2;
        for (int j = 0; j < 96; ++j) {
            const float o = (j < 80) ? sG[r * GS + j] : 0.0f;
            sA[r * AS + j] = (_Float16)((o > 0.0f) ? g * o : 0.0f);
        }
        sRN[r] = fmaxf(fminf(fn, MAXN), MINN);
    }
    __syncthreads();

    // H layer 2
    gemm_f16<104, 96, 3>(sA, Wh2p, sG); __syncthreads();
    ebias_stage(bh2, 104, sEB);         __syncthreads();
    if (r < ROWS) {
        float s2; const float nn = plinear_core(sG, r, 104, sEB, sRN[r], s2);
        const float ls = artanh_c(SQC * nn) / (SQC * nn);
        float pr2 = 0.0f;
        for (int j = 0; j < 104; ++j) { const float o = sG[r * GS + j]; if (o > 0.0f) pr2 += o * o; }
        const float un = fmaxf(ls * s2 * sqrtf(pr2), MINN);
        const float es = tanh_c(SQC * un) / (SQC * un);
        const float fn = es * un;
        const float s3 = (fn > MAXN) ? (MAXN / fn) : 1.0f;
        const float g  = s3 * es * ls * s2;
        for (int j = 0; j < 128; ++j) {
            const float o = (j < 104) ? sG[r * GS + j] : 0.0f;
            sA[r * AS + j] = (_Float16)((o > 0.0f) ? g * o : 0.0f);
        }
        sRN[r] = fmaxf(fminf(fn, MAXN), MINN);
    }
    __syncthreads();

    // H layer 3 (no mob_relu) -> out_h
    gemm_f16<128, 128, 4>(sA, Wh3p, sG); __syncthreads();
    ebias_stage(bh3, 128, sEB);          __syncthreads();
    if (r < ROWS) {
        float s2; const float nn = plinear_core(sG, r, 128, sEB, sRN[r], s2);
        for (int j = 0; j < 128; ++j) {
            const float v = s2 * sG[r * GS + j];
            sH[r * GS + j] = v;
            sA[r * AS + j] = (_Float16)v;
        }
        const float nv = fmaxf(nn, MINN);
        sRN[r] = nv;   // input norm for p_linear(out_h, Wah)
        sNH[r] = nv;   // norm for mobius scalar mul later
    }
    __syncthreads();

    // tan_h = logmap0(p_linear(out_h, Wah, bah))
    gemm_f16<128, 128, 4>(sA, Wahp, sG); __syncthreads();
    ebias_stage(bah, 128, sEB);          __syncthreads();
    if (r < ROWS) {
        float s2; float nn = plinear_core(sG, r, 128, sEB, sRN[r], s2);
        nn = fmaxf(nn, MINN);
        const float ls = artanh_c(SQC * nn) / (SQC * nn);
        for (int j = 0; j < 128; ++j)
            sA[r * AS + j] = (_Float16)(ls * s2 * sG[r * GS + j]);  // tan_h (f16, aliases sA)
    }
    __syncthreads();

    // tan_e = out_e @ Wae^T (+bae added below)
    gemm_f16<128, 128, 4>(sE, Waep, sG); __syncthreads();

    // attention scores: a_e - a_h only needs d = tan_e - tan_h
    if (r < ROWS) {
        float ae = 0.0f, ah = 0.0f;
        for (int j = 0; j < 128; ++j) {
            const float te = sG[r * GS + j] + bae[j];
            const float th = (float)sA[r * AS + j];
            const float d  = 0.5f * (te - th);
            ae += d * Wte[j];
            ah -= d * Wth[j];
        }
        ae += bte[0]; ah += bth[0];
        const float m  = fmaxf(ae, ah);
        const float e0 = expf(ae - m), e1 = expf(ah - m);
        const float w0 = e0 / (e0 + e1);
        sW0[r] = w0; sW1[r] = 1.0f - w0;
    }
    __syncthreads();

    // logmap0(mobius_scalar_mul(w1, out_h)) == fH[r] * out_h  (direction-preserving)
    if (r < ROWS) {
        const float n = sNH[r];
        const float t = tanh_c(sW1[r] * artanh_c(SQC * n));
        const float rs = t / (n * SQC);
        float rn = fabsf(t) / SQC;
        float s2 = 1.0f;
        if (rn > MAXN) { s2 = MAXN / rn; rn = MAXN; }
        const float nn = fmaxf(rn, MINN);
        const float ls = artanh_c(SQC * nn) / (SQC * nn);
        sFH[r] = rs * s2 * ls;
    }
    __syncthreads();

    // pooled partial sums (mean over S=128 -> /128) via f32 atomics
    if (tid < 128) {
        const int j = tid;
        float s = 0.0f;
        for (int rr = 0; rr < ROWS; ++rr) s += sW0[rr] * (float)sE[rr * AS + j];
        atomicAdd(&wsE[b * 128 + j], s * (1.0f / 128.0f));
    } else {
        const int j = tid - 128;
        float s = 0.0f;
        for (int rr = 0; rr < ROWS; ++rr) s += sFH[rr] * sH[rr * GS + j];
        atomicAdd(&wsH[b * 128 + j], s * (1.0f / 128.0f));
    }
}

// ---------------- tiny head: (256 -> 10 relu -> 10) per batch ----------------
__global__ void __launch_bounds__(64)
hypnet_head_kernel(const float* __restrict__ wsE, const float* __restrict__ wsH,
                   const float* __restrict__ Wf1, const float* __restrict__ bf1,
                   const float* __restrict__ Wf2, const float* __restrict__ bf2,
                   float* __restrict__ out)
{
    __shared__ float o[256];
    __shared__ float h1[10];
    const int b = blockIdx.x;
    const int t = threadIdx.x;
    for (int j = t; j < 256; j += 64)
        o[j] = (j < 128) ? wsE[b * 128 + j] : wsH[b * 128 + (j - 128)];
    __syncthreads();
    if (t < 10) {
        float s = bf1[t];
        for (int k = 0; k < 256; ++k) s += Wf1[t * 256 + k] * o[k];
        h1[t] = fmaxf(s, 0.0f);
    }
    __syncthreads();
    if (t < 10) {
        float s = bf2[t];
        for (int k = 0; k < 10; ++k) s += Wf2[t * 10 + k] * h1[k];
        out[b * 10 + t] = s;
    }
}

extern "C" void kernel_launch(void* const* d_in, const int* in_sizes, int n_in,
                              void* d_out, int out_size, void* d_ws, size_t ws_size,
                              hipStream_t stream) {
    (void)in_sizes; (void)n_in; (void)out_size; (void)ws_size;
    const float* x   = (const float*)d_in[0];
    const float* We1 = (const float*)d_in[1];  const float* be1 = (const float*)d_in[2];
    const float* We2 = (const float*)d_in[3];  const float* be2 = (const float*)d_in[4];
    const float* We3 = (const float*)d_in[5];  const float* be3 = (const float*)d_in[6];
    const float* Wh1 = (const float*)d_in[7];  const float* bh1 = (const float*)d_in[8];
    const float* Wh2 = (const float*)d_in[9];  const float* bh2 = (const float*)d_in[10];
    const float* Wh3 = (const float*)d_in[11]; const float* bh3 = (const float*)d_in[12];
    const float* Wae = (const float*)d_in[13]; const float* bae = (const float*)d_in[14];
    const float* Wah = (const float*)d_in[15]; const float* bah = (const float*)d_in[16];
    const float* Wte = (const float*)d_in[17]; const float* bte = (const float*)d_in[18];
    const float* Wth = (const float*)d_in[19]; const float* bth = (const float*)d_in[20];
    const float* Wf1 = (const float*)d_in[21]; const float* bf1 = (const float*)d_in[22];
    const float* Wf2 = (const float*)d_in[23]; const float* bf2 = (const float*)d_in[24];

    float* wsE = (float*)d_ws;               // 1024 x 128
    float* wsH = wsE + 1024 * 128;           // 1024 x 128
    hipMemsetAsync(d_ws, 0, (size_t)1024 * 256 * sizeof(float), stream);

    // padded f16 weight buffers after the pooling accumulators (all 16B aligned)
    _Float16* wpad = (_Float16*)((char*)d_ws + (size_t)1024 * 256 * sizeof(float));
    _Float16* We1p = wpad;                  // 80  x 32  = 2560
    _Float16* We2p = We1p + 80 * 32;        // 112 x 96  = 10752
    _Float16* We3p = We2p + 112 * 96;       // 128 x 128 = 16384
    _Float16* Wh1p = We3p + 128 * 128;      // 80  x 32
    _Float16* Wh2p = Wh1p + 80 * 32;        // 112 x 96
    _Float16* Wh3p = Wh2p + 112 * 96;       // 128 x 128
    _Float16* Wahp = Wh3p + 128 * 128;      // 128 x 128
    _Float16* Waep = Wahp + 128 * 128;      // 128 x 128

    #define CONVW(src, dst, Nout, Kin, Npad, Kpad) \
        convw_kernel<<<((Npad)*(Kpad) + 255) / 256, 256, 0, stream>>>(src, dst, Nout, Kin, Npad, Kpad)
    CONVW(We1, We1p,  80,  32,  80,  32);
    CONVW(We2, We2p, 104,  80, 112,  96);
    CONVW(We3, We3p, 128, 104, 128, 128);
    CONVW(Wh1, Wh1p,  80,  32,  80,  32);
    CONVW(Wh2, Wh2p, 104,  80, 112,  96);
    CONVW(Wh3, Wh3p, 128, 104, 128, 128);
    CONVW(Wah, Wahp, 128, 128, 128, 128);
    CONVW(Wae, Waep, 128, 128, 128, 128);
    #undef CONVW

    hypnet_chunk_kernel<<<4096, 256, 0, stream>>>(
        x, We1p, be1, We2p, be2, We3p, be3,
        Wh1p, bh1, Wh2p, bh2, Wh3p, bh3,
        Waep, bae, Wahp, bah, Wte, bte, Wth, bth,
        wsE, wsH);

    hypnet_head_kernel<<<1024, 64, 0, stream>>>(
        wsE, wsH, Wf1, bf1, Wf2, bf2, (float*)d_out);
}
